// SelfAttention_40759239639534
// MI455X (gfx1250) — matrix operand order repaired and checked by reading, not compiled, not run on hardware
//
#include <hip/hip_runtime.h>
#include <hip/hip_bf16.h>

// ---------------------------------------------------------------------------
// Galerkin self-attention for MI455X (gfx1250), bf16 WMMA pipeline.
// ~51.5 GFLOP of GEMM, ~1 GB HBM traffic -> memory-bound (~45us) when the
// matrix math runs on v_wmma_f32_16x16x32_bf16 and operand staging is done by
// the Tensor Data Mover (tensor_load_to_lds + s_wait_tensorcnt) instead of
// VALU copy loops. All fragment loads are paired ds_load_b128.
// ---------------------------------------------------------------------------

#define B_    8
#define C_    256
#define NH    8
#define DK    32
#define HW    128
#define NPIX  16384   // HW*HW

typedef __attribute__((ext_vector_type(16))) __bf16         v16bf;
typedef __attribute__((ext_vector_type(16))) unsigned short v16us;
typedef __attribute__((ext_vector_type(8)))  float          v8f;
typedef __attribute__((ext_vector_type(4)))  unsigned int   u32x4;
typedef __attribute__((ext_vector_type(8)))  int            i32x8;
typedef __attribute__((ext_vector_type(4)))  int            i32x4;

union Frag { v16us v; uint4 q[2]; };

__device__ __forceinline__ unsigned short f2bf(float f) {
  unsigned int u = __float_as_uint(f);
  u += 0x7FFFu + ((u >> 16) & 1u);          // round-to-nearest-even
  return (unsigned short)(u >> 16);
}

__device__ __forceinline__ v8f wmma_bf16(v16us a, v16us b, v8f c) {
  return __builtin_amdgcn_wmma_f32_16x16x32_bf16(
      false, __builtin_bit_cast(v16bf, a),
      false, __builtin_bit_cast(v16bf, b),
      (short)0, c, false, false);
}

// ---------------------------------------------------------------------------
// TDM: async-load a 2D tile of 2-byte elements into LDS.
//   tile_d0 elements per row (contiguous), tile_d1 rows, row stride stride_d0
//   elements; LDS destination padded 4 DWORDs after every 16 DWORDs, i.e. a
//   64B data row + 16B pad = 80B LDS row stride (matches fragment layout).
// D# fields per CDNA5 ISA.
// ---------------------------------------------------------------------------
__device__ __forceinline__ void tdm_load_2d_bf16(unsigned lds_byte_addr,
                                                 const void* gaddr,
                                                 unsigned tile_d0, unsigned tile_d1,
                                                 unsigned tensor_d0, unsigned tensor_d1,
                                                 unsigned stride_d0) {
  unsigned long long ga = (unsigned long long)gaddr;
  u32x4 g0;
  g0[0] = 1u;                                        // count=1 (valid), flags 0
  g0[1] = lds_byte_addr;                             // lds_addr
  g0[2] = (unsigned)(ga & 0xFFFFFFFFu);              // global_addr[31:0]
  g0[3] = (unsigned)((ga >> 32) & 0x01FFFFFFu)       // global_addr[56:32]
        | (2u << 30);                                // type = 2 ("image")
  i32x8 g1;
  g1[0] = (1 << 16)                                  // data_size = 2 bytes
        | (1 << 20)                                  // pad_enable
        | (3 << 22)                                  // pad_interval: 16 DWORDs
        | (3 << 25);                                 // pad_amount: 4 DWORDs
  g1[1] = (int)((tensor_d0 & 0xFFFFu) << 16);        // tensor_dim0[15:0] @bit48
  g1[2] = (int)(((tensor_d0 >> 16) & 0xFFFFu)        // tensor_dim0[31:16]
        |       ((tensor_d1 & 0xFFFFu) << 16));      // tensor_dim1[15:0] @bit80
  g1[3] = (int)(((tensor_d1 >> 16) & 0xFFFFu)        // tensor_dim1[31:16]
        |       ((tile_d0 & 0xFFFFu) << 16));        // tile_dim0 @bit112
  g1[4] = (int)(tile_d1 & 0xFFFFu);                  // tile_dim1 (tile_dim2=0)
  g1[5] = (int)stride_d0;                            // tensor_dim0_stride[31:0]
  g1[6] = 0;
  g1[7] = 0;
  i32x4 g2 = {0, 0, 0, 0};
  i32x4 g3 = {0, 0, 0, 0};
#if defined(__clang_major__) && (__clang_major__ >= 23)
  i32x8 g4 = {0, 0, 0, 0, 0, 0, 0, 0};
  __builtin_amdgcn_tensor_load_to_lds(g0, g1, g2, g3, g4, 0);
#else
  __builtin_amdgcn_tensor_load_to_lds(g0, g1, g2, g3, 0);
#endif
}

// ---------------------------------------------------------------------------
// K0: DETR sine positional embedding, transposed posT[c][n] f32 (L2-resident,
// 16 MB, re-read per batch by the prep kernel).
// ---------------------------------------------------------------------------
__global__ void ga_pos_kernel(float* __restrict__ posT) {
  int id = blockIdx.x * 256 + threadIdx.x;      // C_*NPIX threads
  int c = id >> 14;
  int n = id & (NPIX - 1);
  int y = n >> 7, xi = n & 127;
  int i = (c < 128) ? c : (c - 128);
  float coord = (c < 128) ? (float)(y + 1) : (float)(xi + 1);
  float norm = coord / (128.0f + 1e-6f) * 6.2831853071795864f;
  float dim_t = powf(10000.0f, (float)(i & ~1) / 128.0f);
  float val = norm / dim_t;
  posT[id] = (i & 1) ? cosf(val) : sinf(val);
}

// K1: pack Wq|Wk|Wv (f32 [co][ci]) -> bf16, same layout.
__global__ void ga_pack_w(const float* __restrict__ Wq,
                          const float* __restrict__ Wk,
                          const float* __restrict__ Wv,
                          unsigned short* __restrict__ Wb) {
  int id = blockIdx.x * 256 + threadIdx.x;      // 3*65536 threads
  const float* src = (id < 65536) ? Wq : (id < 131072 ? Wk : Wv);
  Wb[id] = f2bf(src[id & 65535]);
}

// ---------------------------------------------------------------------------
// K1b: activation prep. LDS-tiled transpose of x[b][c][n] (+posT) into
// pixel-major bf16 buffers qkP[b][n][c] (= x+pos) and xP[b][n][c] (= x).
// ---------------------------------------------------------------------------
__global__ void ga_prep_a(const float* __restrict__ x,
                          const float* __restrict__ posT,
                          unsigned short* __restrict__ qkP,
                          unsigned short* __restrict__ xP) {
  __shared__ float lds_x[64 * 65];
  __shared__ float lds_p[64 * 65];
  const int t = threadIdx.x;
  const int n0 = blockIdx.x * 64, c0 = blockIdx.y * 64, b = blockIdx.z;
#pragma unroll
  for (int it = 0; it < 16; ++it) {
    int idx = it * 256 + t;
    int cl = idx >> 6, nl = idx & 63;           // coalesced along n
    lds_x[cl * 65 + nl] = x[((size_t)b * C_ + c0 + cl) * NPIX + n0 + nl];
    lds_p[cl * 65 + nl] = posT[(size_t)(c0 + cl) * NPIX + n0 + nl];
  }
  __syncthreads();
#pragma unroll
  for (int it = 0; it < 16; ++it) {
    int idx = it * 256 + t;
    int nl = idx >> 6, cl = idx & 63;           // coalesced along c
    float xv = lds_x[cl * 65 + nl];
    float pv = lds_p[cl * 65 + nl];
    size_t o = ((size_t)b * NPIX + n0 + nl) * C_ + c0 + cl;
    xP[o]  = f2bf(xv);
    qkP[o] = f2bf(xv + pv);
  }
}

// ---------------------------------------------------------------------------
// K2: projection GEMM, TDM-fed. Block = 64 pixels x 256 out-channels, K=256
// in 8 steps. Operand tiles are DMA'd into double-buffered LDS by the Tensor
// Data Mover (issued by wave 0, tracked by TENSORcnt), with TDM padding
// reproducing the 80B conflict-free LDS row stride. Wave w owns head w:
// 4 Mtiles x 2 Ntiles of 16x16 f32 accs, fragments as paired ds_load_b128.
// mode 0: Q  | mode 1: K (per-head LN, store transposed) | mode 2: V (LN).
// ---------------------------------------------------------------------------
__global__ void ga_proj_kernel(const unsigned short* __restrict__ Asrc, // [b][n][c] bf16
                               const unsigned short* __restrict__ Wb,   // [co][ci] bf16
                               const float* __restrict__ bias,
                               const float* __restrict__ gamma,
                               const float* __restrict__ beta,
                               unsigned short* __restrict__ outbuf,
                               int mode) {
  __shared__ unsigned short lds_A[2][64 * 40];    // [pix][ci], 80B rows
  __shared__ unsigned short lds_B[2][256 * 40];   // [co][ci],  80B rows

  const int t = threadIdx.x;
  const int wave = t >> 5, lane = t & 31;
  const int b = blockIdx.y;
  const int n0 = blockIdx.x * 64;
  const int head = wave;
  const int lcol = lane & 15, lhalf = lane >> 4;

  const unsigned short* Abase = &Asrc[((size_t)b * NPIX + n0) * C_];

  v8f acc[4][2];
  const v8f vzero = {0.f, 0.f, 0.f, 0.f, 0.f, 0.f, 0.f, 0.f};
#pragma unroll
  for (int ms = 0; ms < 4; ++ms)
#pragma unroll
    for (int ns = 0; ns < 2; ++ns) acc[ms][ns] = vzero;

  if (t < 32) {                                   // wave 0 issues TDM pair kb=0
    tdm_load_2d_bf16((unsigned)(size_t)&lds_A[0][0], Abase, 32, 64, C_, NPIX, C_);
    tdm_load_2d_bf16((unsigned)(size_t)&lds_B[0][0], Wb,    32, 256, C_, C_, C_);
  }

#pragma unroll
  for (int kb = 0; kb < 8; ++kb) {
    if (t < 32) {
      if (kb < 7) {                               // issue kb+1, then wait kb pair
        int nb = (kb + 1) & 1;
        tdm_load_2d_bf16((unsigned)(size_t)&lds_A[nb][0], Abase + (kb + 1) * 32,
                         32, 64, C_, NPIX, C_);
        tdm_load_2d_bf16((unsigned)(size_t)&lds_B[nb][0], Wb + (kb + 1) * 32,
                         32, 256, C_, C_, C_);
        __builtin_amdgcn_s_wait_tensorcnt(2);
      } else {
        __builtin_amdgcn_s_wait_tensorcnt(0);
      }
    }
    __syncthreads();                              // publish buffer kb&1

    const unsigned short* bufA = &lds_A[kb & 1][0];
    const unsigned short* bufB = &lds_B[kb & 1][0];
    // Fragments: A element runs {K=lhalf*8+0..7, K=lhalf*8+16..23};
    //            B element runs {K=lhalf*8+0..7, K=lhalf*8+16..23} -> all paired b128.
    v16us afr[4], bfr[2];
#pragma unroll
    for (int ms = 0; ms < 4; ++ms) {
      Frag f;
      const unsigned short* p = &bufA[(ms * 16 + lcol) * 40 + lhalf * 8];
      f.q[0] = *reinterpret_cast<const uint4*>(p);
      f.q[1] = *reinterpret_cast<const uint4*>(p + 16);
      afr[ms] = f.v;
    }
#pragma unroll
    for (int ns = 0; ns < 2; ++ns) {
      Frag f;
      const unsigned short* p = &bufB[(head * 32 + ns * 16 + lcol) * 40 + lhalf * 8];
      f.q[0] = *reinterpret_cast<const uint4*>(p);
      f.q[1] = *reinterpret_cast<const uint4*>(p + 16);
      bfr[ns] = f.v;
    }
#pragma unroll
    for (int ms = 0; ms < 4; ++ms)
#pragma unroll
      for (int ns = 0; ns < 2; ++ns)
        acc[ms][ns] = wmma_bf16(afr[ms], bfr[ns], acc[ms][ns]);
    __syncthreads();                              // done reading before overwrite
  }

  // ------------------------------ epilogue ------------------------------
  const float bias0 = bias[head * DK + lcol];
  const float bias1 = bias[head * DK + 16 + lcol];
  const size_t bh = (size_t)b * NH + head;

  if (mode == 0) {
#pragma unroll
    for (int ms = 0; ms < 4; ++ms)
#pragma unroll
      for (int r = 0; r < 8; ++r) {
        int row = n0 + ms * 16 + r + (lhalf << 3);
        outbuf[(bh * NPIX + row) * DK + lcol]      = f2bf(acc[ms][0][r] + bias0);
        outbuf[(bh * NPIX + row) * DK + 16 + lcol] = f2bf(acc[ms][1][r] + bias1);
      }
  } else {
    const float g0 = gamma[head * DK + lcol], g1 = gamma[head * DK + 16 + lcol];
    const float e0 = beta[head * DK + lcol],  e1 = beta[head * DK + 16 + lcol];
#pragma unroll
    for (int ms = 0; ms < 4; ++ms)
#pragma unroll
      for (int r = 0; r < 8; ++r) {
        float v0 = acc[ms][0][r] + bias0;
        float v1 = acc[ms][1][r] + bias1;
        float s = v0 + v1, sq = v0 * v0 + v1 * v1;
#pragma unroll
        for (int m = 1; m < 16; m <<= 1) {
          s  += __shfl_xor(s,  m, 16);
          sq += __shfl_xor(sq, m, 16);
        }
        float mu  = s * (1.0f / 32.0f);
        float var = sq * (1.0f / 32.0f) - mu * mu;
        float rs  = rsqrtf(var + 1e-5f);
        float o0 = (v0 - mu) * rs * g0 + e0;
        float o1 = (v1 - mu) * rs * g1 + e1;
        int row = n0 + ms * 16 + r + (lhalf << 3);
        if (mode == 1) {                           // kT: [bh][dk][n]
          outbuf[(bh * DK + lcol) * NPIX + row]      = f2bf(o0);
          outbuf[(bh * DK + 16 + lcol) * NPIX + row] = f2bf(o1);
        } else {                                   // v: [bh][n][dk]
          outbuf[(bh * NPIX + row) * DK + lcol]      = f2bf(o0);
          outbuf[(bh * NPIX + row) * DK + 16 + lcol] = f2bf(o1);
        }
      }
  }
}

__global__ void ga_zero_f32(float* __restrict__ p, int n) {
  int i = blockIdx.x * 256 + threadIdx.x;
  if (i < n) p[i] = 0.0f;
}

// ---------------------------------------------------------------------------
// K3: scores[bh] = k^T v over n=16384, split-K 16-way, f32 atomic reduction.
// lds_K[dk][n] feeds A; lds_V[dk][n] (transposed at staging) feeds B so both
// fragment loads are paired b128s.
// ---------------------------------------------------------------------------
__global__ void ga_scores_kernel(const unsigned short* __restrict__ kT, // [bh][dk][n]
                                 const unsigned short* __restrict__ v,  // [bh][n][dk]
                                 float* __restrict__ scores) {
  __shared__ unsigned short lds_K[32 * 72];     // [dk][n]  (A)
  __shared__ unsigned short lds_V[32 * 72];     // [dk][n]  (B, transposed)
  const int t = threadIdx.x, wave = t >> 5, lane = t & 31;
  const int bh = blockIdx.x;
  const int nbase = blockIdx.y * 1024;
  const int tile = wave & 3, khalf = wave >> 2;
  const int mt = tile >> 1, nt = tile & 1;
  const int lcol = lane & 15, lhalf = lane >> 4;

  const int k_dk = t >> 3, k_ch = t & 7;        // K staging: 1 b128/thread
  const int v_nn = t >> 2, v_ch = t & 3;        // V staging: 1 b128 + 8 scatter

  v8f acc = {0.f, 0.f, 0.f, 0.f, 0.f, 0.f, 0.f, 0.f};
  for (int s = 0; s < 16; ++s) {
    __syncthreads();
    int nb = nbase + s * 64;
    {
      const unsigned short* pk = &kT[((size_t)bh * DK + k_dk) * NPIX + nb + k_ch * 8];
      uint4 vk = *reinterpret_cast<const uint4*>(pk);
      if (s < 15) __builtin_prefetch(pk + 64);
      *reinterpret_cast<uint4*>(&lds_K[k_dk * 72 + k_ch * 8]) = vk;
      Frag fv;
      fv.q[0] = *reinterpret_cast<const uint4*>(
          &v[((size_t)bh * NPIX + nb + v_nn) * DK + v_ch * 8]);
#pragma unroll
      for (int j = 0; j < 8; ++j)               // transpose scatter into LDS
        lds_V[(v_ch * 8 + j) * 72 + v_nn] = ((unsigned short*)&fv.q[0])[j];
    }
    __syncthreads();
    Frag fa, fb;
    {
      const unsigned short* p = &lds_K[(mt * 16 + lcol) * 72 + khalf * 32 + lhalf * 8];
      fa.q[0] = *reinterpret_cast<const uint4*>(p);
      fa.q[1] = *reinterpret_cast<const uint4*>(p + 16);
      const unsigned short* pb = &lds_V[(nt * 16 + lcol) * 72 + khalf * 32 + lhalf * 8];
      fb.q[0] = *reinterpret_cast<const uint4*>(pb);
      fb.q[1] = *reinterpret_cast<const uint4*>(pb + 16);
    }
    acc = wmma_bf16(fa.v, fb.v, acc);
  }
#pragma unroll
  for (int r = 0; r < 8; ++r) {
    int row = mt * 16 + r + (lhalf << 3);
    int col = nt * 16 + lcol;
    atomicAdd(&scores[(size_t)bh * 1024 + row * 32 + col], acc[r]);
  }
}

// K4: scale by 1/n, cast to bf16, and TRANSPOSE to sbT[bh][e][d] so that the
// out-kernel's B fragments (K = d) are contiguous.
__global__ void ga_scores_fin(const float* __restrict__ sf,
                              unsigned short* __restrict__ sbT) {
  int i = blockIdx.x * 256 + threadIdx.x;       // 64*1024
  int bh = i >> 10, d = (i >> 5) & 31, e = i & 31;
  sbT[((size_t)bh << 10) + e * 32 + d] = f2bf(sf[i] * (1.0f / (float)NPIX));
}

// ---------------------------------------------------------------------------
// K5: out = q @ scores per head (K=32, one WMMA step), x2 residual,
// final LN over c=256 via LDS + width-16 shuffles, transposed coalesced store.
// ---------------------------------------------------------------------------
__global__ void ga_out_kernel(const unsigned short* __restrict__ q,   // [bh][n][dk]
                              const unsigned short* __restrict__ sbT, // [bh][e][d]
                              const float* __restrict__ g_ln,
                              const float* __restrict__ b_ln,
                              float* __restrict__ out) {
  __shared__ float lds_O[16 * 260];             // [pix][c] padded
  const int t = threadIdx.x, wave = t >> 5, lane = t & 31;
  const int b = blockIdx.y, n0 = blockIdx.x * 16;
  const int head = wave;
  const int lcol = lane & 15, lhalf = lane >> 4;
  const size_t bh = (size_t)b * NH + head;

  Frag fa, f0, f1;
  {
    const unsigned short* p = &q[(bh * NPIX + n0 + lcol) * DK + lhalf * 8];
    fa.q[0] = *reinterpret_cast<const uint4*>(p);
    fa.q[1] = *reinterpret_cast<const uint4*>(p + 16);
    const unsigned short* p0 = &sbT[(bh << 10) + lcol * 32 + lhalf * 8];
    f0.q[0] = *reinterpret_cast<const uint4*>(p0);
    f0.q[1] = *reinterpret_cast<const uint4*>(p0 + 16);
    const unsigned short* p1 = p0 + 16 * 32;
    f1.q[0] = *reinterpret_cast<const uint4*>(p1);
    f1.q[1] = *reinterpret_cast<const uint4*>(p1 + 16);
  }
  const v8f vzero = {0.f, 0.f, 0.f, 0.f, 0.f, 0.f, 0.f, 0.f};
  v8f c0 = wmma_bf16(fa.v, f0.v, vzero);
  v8f c1 = wmma_bf16(fa.v, f1.v, vzero);
#pragma unroll
  for (int r = 0; r < 8; ++r) {
    int pix = r + (lhalf << 3);
    lds_O[pix * 260 + head * 32 + lcol]      = 2.0f * c0[r];   // out + out
    lds_O[pix * 260 + head * 32 + 16 + lcol] = 2.0f * c1[r];
  }
  __syncthreads();

  int row = t >> 4, sub = t & 15;
  float s = 0.f, sq = 0.f;
#pragma unroll
  for (int j = 0; j < 16; ++j) {
    float vv = lds_O[row * 260 + sub * 16 + j];
    s += vv; sq += vv * vv;
  }
#pragma unroll
  for (int m = 1; m < 16; m <<= 1) {
    s  += __shfl_xor(s,  m, 16);
    sq += __shfl_xor(sq, m, 16);
  }
  float mu  = s * (1.0f / 256.0f);
  float var = sq * (1.0f / 256.0f) - mu * mu;
  float rs  = rsqrtf(var + 1e-5f);
#pragma unroll
  for (int j = 0; j < 16; ++j) {
    int col = sub * 16 + j;
    float vv = lds_O[row * 260 + col];
    lds_O[row * 260 + col] = (vv - mu) * rs * g_ln[col] + b_ln[col];
  }
  __syncthreads();
#pragma unroll
  for (int j = 0; j < 16; ++j) {                // coalesced transposed store
    int col = j * 16 + (t >> 4);
    int pix = t & 15;
    out[((size_t)b * C_ + col) * NPIX + n0 + pix] = lds_O[pix * 260 + col];
  }
}

// ---------------------------------------------------------------------------
extern "C" void kernel_launch(void* const* d_in, const int* in_sizes, int n_in,
                              void* d_out, int out_size, void* d_ws, size_t ws_size,
                              hipStream_t stream) {
  (void)in_sizes; (void)n_in; (void)out_size; (void)ws_size;
  const float* x    = (const float*)d_in[0];
  const float* Wq   = (const float*)d_in[1];
  const float* bq   = (const float*)d_in[2];
  const float* Wk   = (const float*)d_in[3];
  const float* bk   = (const float*)d_in[4];
  const float* Wv   = (const float*)d_in[5];
  const float* bv   = (const float*)d_in[6];
  const float* gK   = (const float*)d_in[7];
  const float* bK   = (const float*)d_in[8];
  const float* gV   = (const float*)d_in[9];
  const float* bV   = (const float*)d_in[10];
  const float* g_ln = (const float*)d_in[11];
  const float* b_ln = (const float*)d_in[12];
  float* out = (float*)d_out;

  char* ws = (char*)d_ws;
  size_t off = 0;
  auto alloc = [&](size_t bytes) -> void* {
    void* p = ws + off;
    off = (off + bytes + 255) & ~(size_t)255;
    return p;
  };
  float*          posT = (float*)alloc((size_t)C_ * NPIX * sizeof(float));        // 16 MB
  unsigned short* Wb   = (unsigned short*)alloc((size_t)3 * C_ * C_ * 2);         // 384 KB
  unsigned short* qkP  = (unsigned short*)alloc((size_t)B_ * NPIX * C_ * 2);      // 64 MB
  unsigned short* xP   = (unsigned short*)alloc((size_t)B_ * NPIX * C_ * 2);      // 64 MB
  unsigned short* qb   = (unsigned short*)alloc((size_t)B_ * NH * NPIX * DK * 2); // 64 MB
  unsigned short* kTb  = (unsigned short*)alloc((size_t)B_ * NH * NPIX * DK * 2); // 64 MB
  unsigned short* vb   = (unsigned short*)alloc((size_t)B_ * NH * NPIX * DK * 2); // 64 MB
  float*          sf   = (float*)alloc((size_t)B_ * NH * DK * DK * sizeof(float));
  unsigned short* sbT  = (unsigned short*)alloc((size_t)B_ * NH * DK * DK * 2);

  ga_pos_kernel<<<(C_ * NPIX) / 256, 256, 0, stream>>>(posT);
  ga_pack_w<<<(3 * C_ * C_) / 256, 256, 0, stream>>>(Wq, Wk, Wv, Wb);

  dim3 gprep(NPIX / 64, C_ / 64, B_);
  ga_prep_a<<<gprep, 256, 0, stream>>>(x, posT, qkP, xP);

  dim3 gproj(NPIX / 64, B_);
  ga_proj_kernel<<<gproj, 256, 0, stream>>>(qkP, Wb,               bq, nullptr, nullptr, qb, 0);
  ga_proj_kernel<<<gproj, 256, 0, stream>>>(qkP, Wb + C_ * C_,     bk, gK, bK, kTb, 1);
  ga_proj_kernel<<<gproj, 256, 0, stream>>>(xP,  Wb + 2 * C_ * C_, bv, gV, bV, vb, 2);

  ga_zero_f32<<<(B_ * NH * DK * DK) / 256, 256, 0, stream>>>(sf, B_ * NH * DK * DK);
  dim3 gsc(B_ * NH, NPIX / 1024);
  ga_scores_kernel<<<gsc, 256, 0, stream>>>(kTb, vb, sf);
  ga_scores_fin<<<(B_ * NH * DK * DK) / 256, 256, 0, stream>>>(sf, sbT);

  dim3 gout(NPIX / 16, B_);
  ga_out_kernel<<<gout, 256, 0, stream>>>(qb, sbT, g_ln, b_ln, out);
}
